// GFlowNetActor_85014582657316
// MI455X (gfx1250) — matrix operand order repaired and checked by reading, not compiled, not run on hardware
//
#include <hip/hip_runtime.h>
#include <hip/hip_bf16.h>
#include <math.h>

// ---------------- problem constants ----------------
#define B_GRAPHS 1024
#define NPG      128
#define EPG      512
#define E_TOT    (B_GRAPHS * EPG)
#define D        64
#define S_STEPS  5
#define MAXST    4
#define GPB      16            // graphs per block (one wave32 per graph)
#define NEGF     (-1e30f)

// ---------------- output layout (floats) ----------------
#define O_ACT    0
#define O_PFSUM  (B_GRAPHS * S_STEPS)                 // 5120
#define O_PFST   (O_PFSUM + B_GRAPHS)                 // 6144
#define O_PBST   (O_PFST + B_GRAPHS * S_STEPS)        // 11264
#define O_USED   (O_PBST + B_GRAPHS * S_STEPS)        // 16384
#define O_STOPN  (O_USED + E_TOT)                     // 540672
#define O_STEPS  (O_STOPN + B_GRAPHS)                 // 541696

// ---------------- dynamic LDS layout (bytes) ----------------
#define OFF_WT    0                          // 8 mats, [n][k] f16: 8*64*64*2 = 65536
#define OFF_VEC   (OFF_WT + 8 * 64 * 64 * 2) // bf,bb,wf,wb,wstop,bu : 6*64*4
#define OFF_STATE (OFF_VEC + 6 * 64 * 4)     // [GPB][64] f32
#define OFF_CF    (OFF_STATE + GPB * 64 * 4) // [GPB][64] f32 (also reused as tmp)
#define OFF_CB    (OFF_CF + GPB * 64 * 4)    // [GPB][64] f32
#define OFF_SROW  (OFF_CB + GPB * 64 * 4)    // [16 waves][16] f32
#define OFF_XVAL  (OFF_SROW + 16 * 16 * 4)   // [GPB][EPG] f32
#define OFF_LIST  (OFF_XVAL + GPB * EPG * 4) // [GPB][EPG] u16
#define OFF_USED  (OFF_LIST + GPB * EPG * 2) // [GPB][16] u32 bitmask
#define OFF_SCAL  (OFF_USED + GPB * 16 * 4)  // scalar arrays
#define SMEM_BYTES (OFF_SCAL + 16 * GPB * 4)

typedef __attribute__((ext_vector_type(16))) _Float16 v16h;
typedef __attribute__((ext_vector_type(8)))  float    v8f;

// ---------------- helpers ----------------
__device__ __forceinline__ v8f wmma_f16(v16h a, v16h b, v8f c) {
  // v_wmma_f32_16x16x32_f16 : D = A(16x32 f16) * B(32x16 f16) + C(16x16 f32)
  return __builtin_amdgcn_wmma_f32_16x16x32_f16(false, a, false, b, (short)0, c,
                                                false, false);
}

// A fragment (16x32 f16): lane t=lane&15 holds row M=t.
// hi=0: halves 0..15 = K kb+0..15
// hi=1: halves 0..15 = K kb+16..31
__device__ __forceinline__ v16h make_A_frag(const float* row, int kb, int lane) {
  int hi = (lane >> 4) & 1;
  int k0 = kb + hi * 16;
  v16h a;
#pragma unroll
  for (int e2 = 0; e2 < 8; ++e2) a[e2] = (_Float16)row[k0 + e2];
#pragma unroll
  for (int e2 = 0; e2 < 8; ++e2) a[8 + e2] = (_Float16)row[k0 + 8 + e2];
  return a;
}

// B fragment (32x16 f16) from LDS weight stored transposed WT[n][k]:
// lane: N = nt*16 + (lane&15), halves e: K = kb + (lane>>4)*16 + e  (contiguous)
__device__ __forceinline__ v16h load_B_frag(const _Float16* WT, int nt, int kb,
                                            int lane) {
  int n  = nt * 16 + (lane & 15);
  int ks = kb + ((lane >> 4) << 4);
  const uint4* q = (const uint4*)(WT + n * 64 + ks);  // 32B aligned
  union { uint4 u[2]; v16h v; } cv;
  cv.u[0] = q[0];
  cv.u[1] = q[1];
  return cv.v;
}

// wave0 batched matvec: out[16][64] = rowA_i @ WT_A (+ rowB_i @ WT_B) + addv
__device__ __forceinline__ void wave0_dual(const float* rowA, const float* rowB,
                                           const _Float16* WT_A,
                                           const _Float16* WT_B,
                                           const float* addv, float* out,
                                           int lane) {
  v16h aA0 = make_A_frag(rowA, 0, lane);
  v16h aA1 = make_A_frag(rowA, 32, lane);
  v16h aB0 = make_A_frag(rowB, 0, lane);
  v16h aB1 = make_A_frag(rowB, 32, lane);
  int t = lane & 15, hi = (lane >> 4) & 1;
#pragma unroll
  for (int nt = 0; nt < 4; ++nt) {
    v8f acc = {};
    acc = wmma_f16(aA0, load_B_frag(WT_A, nt, 0, lane), acc);
    acc = wmma_f16(aA1, load_B_frag(WT_A, nt, 32, lane), acc);
    acc = wmma_f16(aB0, load_B_frag(WT_B, nt, 0, lane), acc);
    acc = wmma_f16(aB1, load_B_frag(WT_B, nt, 32, lane), acc);
    int n = nt * 16 + t;
    float add = addv[n];
#pragma unroll
    for (int r = 0; r < 8; ++r) out[(r + hi * 8) * 64 + n] = acc[r] + add;
  }
}

// per-wave: score a tile of up to 16 gathered edges: x = sum_n relu(ee@W + c[n])*w[n]
__device__ __forceinline__ void score_tile(const unsigned short* lst, int cnt,
                                           int tl, long long gbase,
                                           const float* edge_emb,
                                           const _Float16* WT, const float* cvec,
                                           const float* wvec, float* srow_w,
                                           float* xval_w, int lane) {
  int t = lane & 15, hi = (lane >> 4) & 1;
  int pos = tl * 16 + t;
  int el = (pos < cnt) ? (int)lst[pos] : 0;
  const float* row = edge_emb + (size_t)(gbase + el) * D;
  v16h a0 = make_A_frag(row, 0, lane);
  v16h a1 = make_A_frag(row, 32, lane);
  float part[8];
#pragma unroll
  for (int r = 0; r < 8; ++r) part[r] = 0.f;
#pragma unroll
  for (int nt = 0; nt < 4; ++nt) {
    v8f acc = {};
    acc = wmma_f16(a0, load_B_frag(WT, nt, 0, lane), acc);
    acc = wmma_f16(a1, load_B_frag(WT, nt, 32, lane), acc);
    int n = nt * 16 + t;
    float cv = cvec[n], wv = wvec[n];
#pragma unroll
    for (int r = 0; r < 8; ++r) {
      float h = fmaxf(acc[r] + cv, 0.f);
      part[r] += h * wv;
    }
  }
  // reduce across the 16 lanes of each half (cols)
#pragma unroll
  for (int m = 1; m < 16; m <<= 1)
#pragma unroll
    for (int r = 0; r < 8; ++r) part[r] += __shfl_xor(part[r], m, 32);
  if (t == 0) {
#pragma unroll
    for (int r = 0; r < 8; ++r) srow_w[hi * 8 + r] = part[r];
  }
  // same-wave LDS RAW is kept in order
  int j = lane;
  if (j < 16 && tl * 16 + j < cnt) xval_w[tl * 16 + j] = srow_w[j];
}

__device__ __forceinline__ float logaddexp_(float a, float b) {
  float m = fmaxf(a, b);
  if (m <= -9e29f) return m;
  return m + log1pf(expf(-fabsf(a - b)));
}

__device__ __forceinline__ float gumbel_rng(unsigned a, unsigned b, unsigned c) {
  unsigned h = a * 0x9E3779B1u;
  h ^= (b + 0x85EBCA77u) * 0xC2B2AE3Du;
  h ^= (c + 0x27D4EB2Fu) * 0x165667B1u;
  h ^= h >> 15; h *= 0x2C1B3C6Du; h ^= h >> 12; h *= 0x297A2D39u; h ^= h >> 15;
  float u = ((float)h + 0.5f) * 2.3283064365386963e-10f;  // (0,1)
  return -logf(-logf(u));
}

// ---------------- fused GFlowNet actor kernel ----------------
extern "C" __global__ void __launch_bounds__(512)
gfn_actor_cdna5(const float* __restrict__ node_tokens,
                const float* __restrict__ edge_emb,
                const float* __restrict__ Wf1, const float* __restrict__ Wnf,
                const float* __restrict__ Wf2, const float* __restrict__ bf,
                const float* __restrict__ wf, const float* __restrict__ Wb1,
                const float* __restrict__ Wnb, const float* __restrict__ Wb2,
                const float* __restrict__ bb, const float* __restrict__ wb,
                const float* __restrict__ w_stop,
                const float* __restrict__ b_stop,
                const float* __restrict__ Ws, const float* __restrict__ Wu,
                const float* __restrict__ bu,
                const int* __restrict__ edge_index, float* __restrict__ out) {
  extern __shared__ char sm[];
  _Float16* wT      = (_Float16*)(sm + OFF_WT);  // 8 mats, [n][k]
  float* vbf        = (float*)(sm + OFF_VEC);
  float* vbb        = vbf + 64;
  float* vwf        = vbf + 128;
  float* vwb        = vbf + 192;
  float* vws        = vbf + 256;
  float* vbu        = vbf + 320;
  float* state      = (float*)(sm + OFF_STATE);
  float* cf         = (float*)(sm + OFF_CF);
  float* cb         = (float*)(sm + OFF_CB);
  float* srow       = (float*)(sm + OFF_SROW);
  float* xval       = (float*)(sm + OFF_XVAL);
  unsigned short* list = (unsigned short*)(sm + OFF_LIST);
  unsigned* usedm   = (unsigned*)(sm + OFF_USED);
  int* activeg      = (int*)(sm + OFF_SCAL);
  int* doneA        = activeg + GPB;
  int* stepsA       = activeg + 2 * GPB;
  int* stopnA       = activeg + 3 * GPB;
  int* amaxL        = activeg + 4 * GPB;
  int* amaxG        = activeg + 5 * GPB;
  int* takeE        = activeg + 6 * GPB;
  float* stopl      = (float*)(activeg + 7 * GPB);
  float* pfsum      = (float*)(activeg + 8 * GPB);

  const int tid = threadIdx.x, lane = tid & 31, wv = tid >> 5;
  const int g0 = blockIdx.x * GPB;

  // --- preload weights into LDS as f16, transposed [n][k] ---
  const float* mats[8] = {Wf1, Wnf, Wf2, Wb1, Wnb, Wb2, Ws, Wu};
#pragma unroll
  for (int m = 0; m < 8; ++m) {
    const float* src = mats[m];
    _Float16* dstm = wT + m * 4096;
    for (int idx = tid; idx < 4096; idx += 512) {
      int k = idx >> 6, n = idx & 63;
      dstm[n * 64 + k] = (_Float16)src[idx];
    }
  }
  if (tid < 64) {
    vbf[tid] = bf[tid]; vbb[tid] = bb[tid]; vwf[tid] = wf[tid];
    vwb[tid] = wb[tid]; vws[tid] = w_stop[tid]; vbu[tid] = bu[tid];
  }
  for (int idx = tid; idx < GPB * 64; idx += 512) state[idx] = 0.f;
  for (int idx = tid; idx < GPB * 16; idx += 512) usedm[idx] = 0u;
  if (tid < GPB) {
    activeg[tid] = (g0 + tid) * NPG;
    doneA[tid] = 0; stepsA[tid] = 0; stopnA[tid] = -1;
    pfsum[tid] = 0.f; takeE[tid] = 0; amaxG[tid] = 0; amaxL[tid] = 0;
  }
  __syncthreads();
  const float bstop = b_stop[0];

  for (int stp = 0; stp < S_STEPS; ++stp) {
    // --- phase A (wave0): cf = state@Wf2 + nt[active]@Wnf + bf ; stop logits ---
    if (wv == 0) {
      int t = lane & 15;
      const float* rowA = state + t * 64;
      const float* rowB = node_tokens + (size_t)activeg[t] * D;
      wave0_dual(rowA, rowB, wT + 2 * 4096, wT + 1 * 4096, vbf, cf, lane);
      if (lane < GPB) {
        float sa = bstop;
        for (int k = 0; k < 64; ++k) sa += state[lane * 64 + k] * vws[k];
        stopl[lane] = sa;
      }
    }
    __syncthreads();

    // --- phase B (per wave = per graph): forward score + gumbel sampling ---
    {
      int i = wv, gid = g0 + i;
      long long gbase = (long long)gid * EPG;
      int donePrev = doneA[i];
      if (!donePrev) {
        int act = activeg[i];
        unsigned short* lst = list + i * EPG;
        int cnt = 0;
#pragma unroll
        for (int j = 0; j < 16; ++j) {
          int e = lane + 32 * j;
          int srcn = edge_index[gbase + e];
          unsigned uw = usedm[i * 16 + j];
          bool v = (srcn == act) && !((uw >> lane) & 1u);
          unsigned msk = (unsigned)__ballot(v);
          if (v) lst[cnt + __popc(msk & ((1u << lane) - 1u))] = (unsigned short)e;
          cnt += __popc(msk);
        }
        int ntile = (cnt + 15) >> 4;
        for (int tl = 0; tl < ntile; ++tl)
          score_tile(lst, cnt, tl, gbase, edge_emb, wT + 0, cf + i * 64, vwf,
                     srow + i * 16, xval + i * EPG, lane);
        float lm = NEGF;
        for (int p = lane; p < cnt; p += 32) lm = fmaxf(lm, xval[i * EPG + p]);
#pragma unroll
        for (int m = 1; m < 32; m <<= 1) lm = fmaxf(lm, __shfl_xor(lm, m, 32));
        float ssum = 0.f;
        for (int p = lane; p < cnt; p += 32) ssum += expf(xval[i * EPG + p] - lm);
#pragma unroll
        for (int m = 1; m < 32; m <<= 1) ssum += __shfl_xor(ssum, m, 32);
        float lse_e = (cnt > 0) ? lm + logf(fmaxf(ssum, 1e-30f)) : NEGF;
        int allow = (stepsA[i] < MAXST);
        float st = allow ? stopl[i] : NEGF;
        float Z = logaddexp_(lse_e, st);
        // gumbel-max over valid edges (tie -> min index)
        float bvv = NEGF;
        int bp = 1 << 30;
        for (int p = lane; p < cnt; p += 32) {
          int el = lst[p];
          float gn = gumbel_rng((unsigned)gid, (unsigned)stp, (unsigned)el);
          float se = xval[i * EPG + p] - Z + gn;
          if (se > bvv || (se == bvv && p < bp)) { bvv = se; bp = p; }
        }
#pragma unroll
        for (int m = 1; m < 32; m <<= 1) {
          float ov = __shfl_xor(bvv, m, 32);
          int op = __shfl_xor(bp, m, 32);
          if (ov > bvv || (ov == bvv && op < bp)) { bvv = ov; bp = op; }
        }
        if (lane == 0) {
          float ss = allow ? (st - Z) + gumbel_rng((unsigned)gid, (unsigned)stp,
                                                   0x00FFFFFFu)
                           : NEGF;
          bool hasE = cnt > 0;
          bool chooseStop = (ss > bvv) || !hasE;
          float lpstop = allow ? (stopl[i] - Z) : 0.f;
          float lpf;
          int actOut;
          if (chooseStop) {
            lpf = lpstop;
            actOut = -1;
            takeE[i] = 0;
            if (stopnA[i] < 0) stopnA[i] = activeg[i] - gid * NPG;
            doneA[i] = 1;
            amaxG[i] = 0; amaxL[i] = 0;
          } else {
            int el = lst[bp];
            lpf = xval[i * EPG + bp] - Z;
            actOut = (int)(gbase + el);
            takeE[i] = 1;
            amaxL[i] = el;
            amaxG[i] = (int)(gbase + el);
            usedm[i * 16 + (el >> 5)] |= (1u << (el & 31));
            stepsA[i] += 1;
            activeg[i] = edge_index[E_TOT + gbase + el];  // new active = edst
          }
          out[O_ACT + gid * S_STEPS + stp] = (float)actOut;
          out[O_PFST + gid * S_STEPS + stp] = lpf;
          pfsum[i] += lpf;
        }
      } else if (lane == 0) {
        takeE[i] = 0;
        amaxG[i] = 0;
        out[O_ACT + gid * S_STEPS + stp] = -1.f;
        out[O_PFST + gid * S_STEPS + stp] = 0.f;
      }
    }
    __syncthreads();

    // --- phase C (wave0): state update + cb = state_new@Wb2 + nt[new]@Wnb + bb ---
    if (wv == 0) {
      int t = lane & 15;
      const float* rowA = state + t * 64;
      const float* rowB = edge_emb + (size_t)amaxG[t] * D;  // rel
      wave0_dual(rowA, rowB, wT + 6 * 4096, wT + 7 * 4096, vbu, cf, lane);  // tmp in cf
      for (int idx = lane; idx < GPB * 64; idx += 32) {
        int i2 = idx >> 6;
        if (takeE[i2]) state[idx] = tanhf(cf[idx]);
      }
      const float* rowC = state + t * 64;  // updated state (same-wave order OK)
      const float* rowDn = node_tokens + (size_t)activeg[t] * D;
      wave0_dual(rowC, rowDn, wT + 5 * 4096, wT + 4 * 4096, vbb, cb, lane);
    }
    __syncthreads();

    // --- phase D (per wave): backward log-prob over incoming edges ---
    {
      int i = wv, gid = g0 + i;
      long long gbase = (long long)gid * EPG;
      float lpb = 0.f;
      if (takeE[i]) {
        int act = activeg[i];
        unsigned short* lst = list + i * EPG;
        int cnt = 0;
#pragma unroll
        for (int j = 0; j < 16; ++j) {
          int e = lane + 32 * j;
          bool v = (edge_index[E_TOT + gbase + e] == act);
          unsigned msk = (unsigned)__ballot(v);
          if (v) lst[cnt + __popc(msk & ((1u << lane) - 1u))] = (unsigned short)e;
          cnt += __popc(msk);
        }
        int ntile = (cnt + 15) >> 4;
        for (int tl = 0; tl < ntile; ++tl)
          score_tile(lst, cnt, tl, gbase, edge_emb, wT + 3 * 4096, cb + i * 64,
                     vwb, srow + i * 16, xval + i * EPG, lane);
        float lm = NEGF;
        for (int p = lane; p < cnt; p += 32) lm = fmaxf(lm, xval[i * EPG + p]);
#pragma unroll
        for (int m = 1; m < 32; m <<= 1) lm = fmaxf(lm, __shfl_xor(lm, m, 32));
        float ssum = 0.f;
        for (int p = lane; p < cnt; p += 32) ssum += expf(xval[i * EPG + p] - lm);
#pragma unroll
        for (int m = 1; m < 32; m <<= 1) ssum += __shfl_xor(ssum, m, 32);
        float lse_b = lm + logf(fmaxf(ssum, 1e-30f));
        // xb at the chosen edge (guaranteed in list since edst[amax]==act)
        int al = amaxL[i];
        float xsel = NEGF;
        for (int p = lane; p < cnt; p += 32)
          if ((int)lst[p] == al) xsel = xval[i * EPG + p];
#pragma unroll
        for (int m = 1; m < 32; m <<= 1)
          xsel = fmaxf(xsel, __shfl_xor(xsel, m, 32));
        lpb = xsel - lse_b;
      }
      if (lane == 0) out[O_PBST + gid * S_STEPS + stp] = lpb;
    }
    __syncthreads();
  }

  // --- epilogue ---
  {
    int i = wv, gid = g0 + i;
    long long gbase = (long long)gid * EPG;
    if (lane == 0) {
      out[O_PFSUM + gid] = pfsum[i];
      out[O_STEPS + gid] = (float)stepsA[i];
      int sn = stopnA[i];
      out[O_STOPN + gid] = (float)(sn < 0 ? activeg[i] - gid * NPG : sn);
    }
    for (int e = lane; e < EPG; e += 32) {
      int bit = (usedm[i * 16 + (e >> 5)] >> (e & 31)) & 1;
      out[O_USED + gbase + e] = (float)bit;
    }
  }
}

// ---------------- host launcher ----------------
extern "C" void kernel_launch(void* const* d_in, const int* in_sizes, int n_in,
                              void* d_out, int out_size, void* d_ws,
                              size_t ws_size, hipStream_t stream) {
  (void)in_sizes; (void)n_in; (void)d_ws; (void)ws_size; (void)out_size;
  const float* node_tokens = (const float*)d_in[0];
  const float* edge_emb    = (const float*)d_in[1];
  const float* Wf1   = (const float*)d_in[2];
  const float* Wnf   = (const float*)d_in[3];
  const float* Wf2   = (const float*)d_in[4];
  const float* bf    = (const float*)d_in[5];
  const float* wf    = (const float*)d_in[6];
  const float* Wb1   = (const float*)d_in[7];
  const float* Wnb   = (const float*)d_in[8];
  const float* Wb2   = (const float*)d_in[9];
  const float* bb    = (const float*)d_in[10];
  const float* wb    = (const float*)d_in[11];
  const float* w_stop = (const float*)d_in[12];
  const float* b_stop = (const float*)d_in[13];
  const float* Ws    = (const float*)d_in[14];
  const float* Wu    = (const float*)d_in[15];
  const float* bu    = (const float*)d_in[16];
  const int* edge_index = (const int*)d_in[17];
  float* out = (float*)d_out;

  dim3 grid(B_GRAPHS / GPB);  // 64 blocks
  dim3 block(512);            // 16 wave32s
  gfn_actor_cdna5<<<grid, block, SMEM_BYTES, stream>>>(
      node_tokens, edge_emb, Wf1, Wnf, Wf2, bf, wf, Wb1, Wnb, Wb2, bb, wb,
      w_stop, b_stop, Ws, Wu, bu, edge_index, out);
}
